// PySLSTMLayer_5403068858721
// MI455X (gfx1250) — compile-verified, matrix operand order checked by reading, not run on hardware
//
#include <hip/hip_runtime.h>
#include <hip/hip_bf16.h>

// ---- CDNA5 WMMA types -------------------------------------------------------
typedef __attribute__((ext_vector_type(16))) __bf16 v16bf;
typedef __attribute__((ext_vector_type(8)))  __bf16 v8bf;
typedef __attribute__((ext_vector_type(8)))  float  v8f;

// Problem constants (from reference)
constexpr int BATCH = 32768;
constexpr int IN    = 512;
constexpr int HID   = 512;
constexpr int BM    = 128;           // batch rows per block (2 row-tiles / wave)
constexpr int BN    = 64;            // h-columns per block (x4 gates internally)
constexpr int KC    = 32;            // K chunk per LDS stage (one WMMA K-step)
constexpr int LDS_STRIDE = 40;       // padded row stride in bf16 (mult of 8)

// Load one 16x32 bf16 WMMA operand fragment from LDS.
// ISA layout (16-bit operand, 16x32): lanes 0-15 hold K={0..7,16..23},
// lanes 16-31 hold K={8..15,24..31}. p0 = &row[kof], kof = (lane<16)?0:8.
__device__ __forceinline__ v16bf ld_frag(const __bf16* p0) {
    v8bf lo = *reinterpret_cast<const v8bf*>(p0);        // K kof..kof+7
    v8bf hi = *reinterpret_cast<const v8bf*>(p0 + 16);   // K 16+kof..23+kof
    return __builtin_shufflevector(lo, hi, 0,1,2,3,4,5,6,7,8,9,10,11,12,13,14,15);
}

__device__ __forceinline__ void cvt_store4(__bf16* dst, float4 v) {
    __bf16 t[4] = { (__bf16)v.x, (__bf16)v.y, (__bf16)v.z, (__bf16)v.w };
    *reinterpret_cast<unsigned long long*>(dst) =
        *reinterpret_cast<unsigned long long*>(t);
}

// Per-thread register staging for one K-chunk: A 4 x float4, B 8 x float4.
struct Stage {
    float4 a[4];
    float4 b[8];
};

// Issue all global loads for one K-chunk (no waits; loads stay in flight).
__device__ __forceinline__ void stage_fetch(Stage& st,
                                            const float* __restrict__ aSrc,
                                            const float* __restrict__ bSrc,
                                            int rowBase, int colBase,
                                            int kcol, int tid) {
    #pragma unroll
    for (int j = 0; j < 4; ++j) {
        const int u  = tid + j * 256;     // 0..1023
        const int r  = u >> 3;            // 0..127
        const int k4 = (u & 7) * 4;       // 0..28
        const float* p = aSrc + (long)(rowBase + r) * IN + kcol + k4;
        st.a[j] = *reinterpret_cast<const float4*>(p);
        __builtin_prefetch(p + 2 * KC, 0, 1);   // speculative, 2 chunks ahead
    }
    #pragma unroll
    for (int j = 0; j < 8; ++j) {
        const int u   = tid + j * 256;    // 0..2047
        const int col = u >> 3;           // 0..255 = gate*64 + cg
        const int k4  = (u & 7) * 4;
        const int g   = col >> 6;
        const int cg  = col & 63;
        const float* p = bSrc + (long)(g * HID + colBase + cg) * IN + kcol + k4;
        st.b[j] = *reinterpret_cast<const float4*>(p);
        __builtin_prefetch(p + 2 * KC, 0, 1);
    }
}

// Convert staged f32 -> bf16 and write to LDS (compiler inserts the one wait).
__device__ __forceinline__ void stage_commit(const Stage& st,
                                             __bf16* __restrict__ dA,
                                             __bf16* __restrict__ dB,
                                             int tid) {
    #pragma unroll
    for (int j = 0; j < 4; ++j) {
        const int u  = tid + j * 256;
        const int r  = u >> 3;
        const int k4 = (u & 7) * 4;
        cvt_store4(&dA[r * LDS_STRIDE + k4], st.a[j]);
    }
    #pragma unroll
    for (int j = 0; j < 8; ++j) {
        const int u   = tid + j * 256;
        const int col = u >> 3;
        const int k4  = (u & 7) * 4;
        cvt_store4(&dB[col * LDS_STRIDE + k4], st.b[j]);
    }
}

__global__ __launch_bounds__(256)
void slstm_fused_wmma(const float* __restrict__ x,
                      const float* __restrict__ h_prev,
                      const float* __restrict__ c_prev,
                      const float* __restrict__ n_prev,
                      const float* __restrict__ W,
                      const float* __restrict__ R,
                      const float* __restrict__ bias,
                      float* __restrict__ out_h,
                      float* __restrict__ out_c,
                      float* __restrict__ out_n) {
    __shared__ __bf16 lA[2][BM * LDS_STRIDE];        // 20.0 KB  [buf][row][k]
    __shared__ __bf16 lB[2][4 * BN * LDS_STRIDE];    // 40.0 KB  [buf][gate*64+col][k]

    const int tid  = threadIdx.x;
    const int lane = tid & 31;
    const int wave = tid >> 5;

    const int rowBase = blockIdx.x * BM;    // 256 blocks
    const int colBase = blockIdx.y * BN;    // 8 blocks

    const int waveRow = (wave & 3) * 16;    // 0,16,32,48 (+64 for 2nd row-tile)
    const int waveCol = (wave >> 2) * 32;   // 0,32

    v8f acc[2][4][2] = {};                  // [row-tile][gate][col-tile]

    const int lhalf = (lane < 16) ? 0 : 8;  // K sub-offset per ISA layout
    const int lmod  = lane & 15;

    // One K-step of compute from the "current" buffers: 10 frag loads, 16 WMMAs.
    auto compute = [&](const __bf16* cA, const __bf16* cB) {
        const v16bf a0 = ld_frag(&cA[(waveRow + lmod) * LDS_STRIDE + lhalf]);
        const v16bf a1 = ld_frag(&cA[(waveRow + 64 + lmod) * LDS_STRIDE + lhalf]);
        #pragma unroll
        for (int i = 0; i < 8; ++i) {
            const int g = i >> 1, t = i & 1;
            const int col = g * BN + waveCol + t * 16 + lmod;
            const v16bf bf_ = ld_frag(&cB[col * LDS_STRIDE + lhalf]);
            acc[0][g][t] = __builtin_amdgcn_wmma_f32_16x16x32_bf16(
                false, a0, false, bf_, (short)0, acc[0][g][t], false, false);
            acc[1][g][t] = __builtin_amdgcn_wmma_f32_16x16x32_bf16(
                false, a1, false, bf_, (short)0, acc[1][g][t], false, false);
        }
    };

    __bf16 *Ac = lA[0], *An = lA[1], *Bc = lB[0], *Bn = lB[1];
    Stage st;

    // Prime buffer with chunk 0 (x/W half).
    stage_fetch(st, x, W, rowBase, colBase, 0, tid);
    stage_commit(st, Ac, Bc, tid);
    __syncthreads();

    // Pipeline: fetch k+1 -> compute k (hides HBM latency) -> commit -> barrier.
    // Phase 1: chunks 1..15 sourced from x/W (uniform pointers, no selects).
    for (int kn = KC; kn < IN; kn += KC) {
        stage_fetch(st, x, W, rowBase, colBase, kn, tid);
        compute(Ac, Bc);
        stage_commit(st, An, Bn, tid);
        __syncthreads();
        __bf16* s;
        s = Ac; Ac = An; An = s;
        s = Bc; Bc = Bn; Bn = s;
    }
    // Bridge: fetch first h_prev/R chunk while computing last x/W chunk.
    {
        stage_fetch(st, h_prev, R, rowBase, colBase, 0, tid);
        compute(Ac, Bc);
        stage_commit(st, An, Bn, tid);
        __syncthreads();
        __bf16* s;
        s = Ac; Ac = An; An = s;
        s = Bc; Bc = Bn; Bn = s;
    }
    // Phase 2: chunks 1..15 sourced from h_prev/R.
    for (int kn = KC; kn < HID; kn += KC) {
        stage_fetch(st, h_prev, R, rowBase, colBase, kn, tid);
        compute(Ac, Bc);
        stage_commit(st, An, Bn, tid);
        __syncthreads();
        __bf16* s;
        s = Ac; Ac = An; An = s;
        s = Bc; Bc = Bn; Bn = s;
    }
    // Tail: compute last chunk (no further staging, no barrier needed).
    compute(Ac, Bc);

    // ---- fused sLSTM epilogue ---------------------------------------------
    // C/D layout: VGPR v, lane l -> M = v + (l>=16 ? 8 : 0), N = l & 15.
    #pragma unroll
    for (int rt = 0; rt < 2; ++rt) {
        const int rTile = rowBase + waveRow + rt * 64 + lhalf;
        #pragma unroll
        for (int t = 0; t < 2; ++t) {
            const int gcol = colBase + waveCol + t * 16 + lmod;
            const float bz = bias[gcol];
            const float bi = bias[HID + gcol];
            const float bf = bias[2 * HID + gcol];
            const float bo = bias[3 * HID + gcol];
            #pragma unroll
            for (int v = 0; v < 8; ++v) {
                const long row = rTile + v;
                const long idx = row * HID + gcol;

                const float pz = acc[rt][0][t][v] + bz;
                const float pi = acc[rt][1][t][v] + bi;
                const float pf = acc[rt][2][t][v] + bf;
                const float po = acc[rt][3][t][v] + bo;

                const float e2z = __expf(2.0f * pz);          // tanh via exp
                const float z   = (e2z - 1.0f) / (e2z + 1.0f);
                const float ig  = __expf(pi);
                const float fg  = __expf(pf);
                const float og  = 1.0f / (1.0f + __expf(-po)); // sigmoid

                const float cp = c_prev[idx];
                const float np = n_prev[idx];
                const float c  = fg * cp + ig * z;
                const float n  = fg * np + ig;
                const float h  = og * (c / n);

                out_h[idx] = h;
                out_c[idx] = c;
                out_n[idx] = n;
            }
        }
    }
}

extern "C" void kernel_launch(void* const* d_in, const int* in_sizes, int n_in,
                              void* d_out, int out_size, void* d_ws, size_t ws_size,
                              hipStream_t stream) {
    (void)in_sizes; (void)n_in; (void)out_size; (void)d_ws; (void)ws_size;
    const float* x      = (const float*)d_in[0];
    const float* h_prev = (const float*)d_in[1];
    const float* c_prev = (const float*)d_in[2];
    const float* n_prev = (const float*)d_in[3];
    const float* W      = (const float*)d_in[4];
    const float* R      = (const float*)d_in[5];
    const float* b      = (const float*)d_in[6];

    float* out   = (float*)d_out;
    const long BH = (long)BATCH * HID;   // 16,777,216
    float* out_h = out;
    float* out_c = out + BH;
    float* out_n = out + 2 * BH;

    dim3 grid(BATCH / BM, HID / BN);     // 256 x 8
    dim3 block(256);                     // 8 waves (wave32)
    slstm_fused_wmma<<<grid, block, 0, stream>>>(
        x, h_prev, c_prev, n_prev, W, R, b, out_h, out_c, out_n);
}
